// TransformerBBSparse_59665685676430
// MI455X (gfx1250) — compile-verified, matrix operand order checked by reading, not run on hardware
//
#include <hip/hip_runtime.h>
#include <math.h>

typedef __attribute__((ext_vector_type(16))) _Float16 v16h;
typedef __attribute__((ext_vector_type(8)))  float    v8f;

#define WMMA(a, b, c) __builtin_amdgcn_wmma_f32_16x16x32_f16( \
    false, (a), false, (b), (short)0, (c), false, false)

// ---------------------------------------------------------------------------
// WMMA fragment helpers (wave32). ISA 7.12.2 layouts:
//  A (16x32 f16): lane l holds row r=l%16; half j of VGPR v covers
//      k = (v<4?0:16) + (l/16)*8 + 2*(v&3) + (j&1)
//  B (32x16 f16): lane l holds col c=l%16; same K pattern, element (k,c).
//  C/D (16x16 f32): lane l holds col c=l%16, rows (l/16)*8 + v.
//
// swz(r,k): inverse A mapping -> fragment-swizzled LDS tile [32 lanes][16 halfs]
// so an A fragment is two contiguous ds_load_b128 per lane.
// ---------------------------------------------------------------------------
__device__ inline int swz(int r, int k) {
  return (((r & 15) + (((k >> 3) & 1) << 4)) << 4) + (((k >> 4) & 1) << 3) + (k & 7);
}

__device__ inline v16h fragA_swz(const _Float16* base) {
  int lane = threadIdx.x & 31;
  return *(const v16h*)(base + lane * 16);
}

template <typename T>
__device__ inline v16h fragA(const T* src, int ld, int rmax, int kmax) {
  int lane = threadIdx.x & 31;
  int r = lane & 15, half = lane >> 4;
  bool rok = (r < rmax);
  v16h a;
#pragma unroll
  for (int v = 0; v < 8; ++v) {
    int k0 = ((v & 4) << 2) + half * 8 + ((v & 3) << 1);
    a[2 * v]     = (rok && (k0     < kmax)) ? (_Float16)src[r * ld + k0]     : (_Float16)0.f;
    a[2 * v + 1] = (rok && (k0 + 1 < kmax)) ? (_Float16)src[r * ld + k0 + 1] : (_Float16)0.f;
  }
  return a;
}

template <typename T>
__device__ inline v16h fragB(const T* src, int ld, int kmax, int cmax) {
  int lane = threadIdx.x & 31;
  int c = lane & 15, half = lane >> 4;
  bool cok = (c < cmax);
  v16h b;
#pragma unroll
  for (int v = 0; v < 8; ++v) {
    int k0 = ((v & 4) << 2) + half * 8 + ((v & 3) << 1);
    b[2 * v]     = (cok && (k0     < kmax)) ? (_Float16)src[k0 * ld + c]       : (_Float16)0.f;
    b[2 * v + 1] = (cok && (k0 + 1 < kmax)) ? (_Float16)src[(k0 + 1) * ld + c] : (_Float16)0.f;
  }
  return b;
}

__device__ inline float gelu_erf(float z) {
  return 0.5f * z * (1.f + erff(z * 0.70710678118654752f));
}

// ---------------------------------------------------------------------------
// 1. embed: mem[b,s,:] = x[b,s]*in_w + in_b + pe[s,:]
// ---------------------------------------------------------------------------
__global__ void k_embed(const float* __restrict__ x, const float* __restrict__ in_w,
                        const float* __restrict__ in_b, const float* __restrict__ pe,
                        float* __restrict__ mem, int rows) {
  int i = blockIdx.x * blockDim.x + threadIdx.x;
  if (i >= rows * 64) return;
  int e = i & 63, row = i >> 6, s = row % 2560;
  mem[i] = x[row] * in_w[e] + in_b[e] + pe[s * 64 + e];
}

// ---------------------------------------------------------------------------
// 2. TBatchNorm stats, deterministic two-pass (sum, sumsq per feature)
// ---------------------------------------------------------------------------
__global__ void k_stats_partial(const float* __restrict__ X, int rows, int G,
                                float* __restrict__ partial) {
  __shared__ float ssum[256], ssq[256];
  int f = threadIdx.x & 63, sub = threadIdx.x >> 6;
  int chunk = (rows + G - 1) / G;
  int r0 = blockIdx.x * chunk;
  int r1 = r0 + chunk; if (r1 > rows) r1 = rows;
  float s = 0.f, q = 0.f;
  for (int r = r0 + sub; r < r1; r += 4) { float v = X[r * 64 + f]; s += v; q += v * v; }
  ssum[threadIdx.x] = s; ssq[threadIdx.x] = q;
  __syncthreads();
  if (sub == 0) {
    for (int j = 1; j < 4; ++j) { s += ssum[f + 64 * j]; q += ssq[f + 64 * j]; }
    partial[blockIdx.x * 128 + f] = s;
    partial[blockIdx.x * 128 + 64 + f] = q;
  }
}

__global__ void k_stats_final(const float* __restrict__ partial, int G, float invR,
                              float* __restrict__ meanv, float* __restrict__ rstdv) {
  int f = threadIdx.x;
  if (f >= 64) return;
  float s = 0.f, q = 0.f;
  for (int g = 0; g < G; ++g) { s += partial[g * 128 + f]; q += partial[g * 128 + 64 + f]; }
  float m = s * invR;
  float v = q * invR - m * m;                 // biased variance (torch BN normalization)
  meanv[f] = m;
  rstdv[f] = rsqrtf(v + 1e-5f);
}

// ---------------------------------------------------------------------------
// 3. Fused self-attention over a T-token block (T=40 enc / T=24 dec, pad to 48)
//    X += OutProj( softmax(QK^T/sqrt(dh)) V )  with Q,K,V from tbn(X).
//    One workgroup per (batch, block). 8 waves, 156 WMMA per WG.
// ---------------------------------------------------------------------------
__global__ __launch_bounds__(256) void k_attn(
    float* __restrict__ X, int nb, int T,
    const float* __restrict__ meanv, const float* __restrict__ rstdv,
    const float* __restrict__ gam, const float* __restrict__ bet,
    const float* __restrict__ Wqkv, const float* __restrict__ bqkv,
    const float* __restrict__ Wo, const float* __restrict__ bo) {
  __shared__ _Float16 qkv[48 * 192];                          // Q|K|V token rows
  __shared__ float    sc[48 * 96];                            // scores, 2 heads
  __shared__ float    obuf[48 * 64] __attribute__((aligned(32)));
  // front 6KB of obuf aliased as fragment-swizzled xn: 6 tiles (3 mt x 2 ks)
  _Float16* xn = (_Float16*)obuf;

  int b = blockIdx.x / nb, blk = blockIdx.x % nb;
  int row0 = (b * nb + blk) * T;
  int tid = threadIdx.x, lane = tid & 31, wv = tid >> 5;

  for (int i = tid; i < 48 * 64; i += 256) {
    int r = i >> 6, e = i & 63;
    float v = 0.f;
    if (r < T) v = (X[(row0 + r) * 64 + e] - meanv[e]) * rstdv[e] * gam[e] + bet[e];
    xn[((r >> 4) * 2 + (e >> 5)) * 512 + swz(r & 15, e & 31)] = (_Float16)v;
  }
  __syncthreads();

  // QKV = xn[48x64] @ Wqkv[64x192] + bqkv : 3 Mtiles x 12 Ntiles x 2 ksteps
  for (int t = wv; t < 36; t += 8) {
    int mt = t / 12, nt = t % 12;
    v8f acc = {};
#pragma unroll
    for (int ks = 0; ks < 2; ++ks) {
      v16h a  = fragA_swz(xn + (mt * 2 + ks) * 512);
      v16h bb = fragB(Wqkv + ks * 32 * 192 + nt * 16, 192, 32, 16);
      acc = WMMA(a, bb, acc);
    }
    int c = lane & 15, r0 = (lane >> 4) * 8;
#pragma unroll
    for (int v = 0; v < 8; ++v) {
      int r = mt * 16 + r0 + v, cc = nt * 16 + c;
      qkv[r * 192 + cc] = (_Float16)(acc[v] + bqkv[cc]);
    }
  }
  __syncthreads();

  for (int hp = 0; hp < 2; ++hp) {            // heads {0,1} then {2,3}
    // scores S = Q K^T * 0.25 : 3 Mtiles x (2 heads x 3 keytiles)
    for (int t = wv; t < 18; t += 8) {
      int mt = t / 6, q6 = t % 6, hh = q6 / 3, nt = q6 % 3;
      int h = hp * 2 + hh;
      v16h a  = fragA(qkv + (mt * 16) * 192 + h * 16, 192, 16, 16);           // Q tile
      v16h bb = fragA(qkv + (nt * 16) * 192 + 64 + h * 16, 192, 16, 16);      // K^T tile
      v8f acc = {};
      acc = WMMA(a, bb, acc);
      int c = lane & 15, r0 = (lane >> 4) * 8;
#pragma unroll
      for (int v = 0; v < 8; ++v)
        sc[(mt * 16 + r0 + v) * 96 + hh * 48 + nt * 16 + c] = acc[v] * 0.25f;
    }
    __syncthreads();

    // softmax rows, zero padded rows/cols (96 row-slots for the 2 heads)
    if (tid < 96) {
      int hh = tid / 48, r = tid % 48;
      float* srow = sc + r * 96 + hh * 48;
      if (r < T) {
        float mx = -1e30f;
        for (int c2 = 0; c2 < T; ++c2) mx = fmaxf(mx, srow[c2]);
        float sum = 0.f;
        for (int c2 = 0; c2 < T; ++c2) { float e = __expf(srow[c2] - mx); srow[c2] = e; sum += e; }
        float inv = 1.f / sum;
        for (int c2 = 0; c2 < T; ++c2) srow[c2] *= inv;
        for (int c2 = T; c2 < 48; ++c2) srow[c2] = 0.f;
      } else {
        for (int c2 = 0; c2 < 48; ++c2) srow[c2] = 0.f;
      }
    }
    __syncthreads();

    // O = P V : 3 Mtiles x 2 heads, K = 48 padded keys (2 ksteps)
    for (int t = wv; t < 6; t += 8) {
      int mt = t / 2, hh = t % 2, h = hp * 2 + hh;
      v8f acc = {};
#pragma unroll
      for (int ks = 0; ks < 2; ++ks) {
        int kmax = 48 - ks * 32;               // 32 then 16 valid K
        v16h a  = fragA(sc + (mt * 16) * 96 + hh * 48 + ks * 32, 96, 16, kmax);
        v16h bb = fragB(qkv + (ks * 32) * 192 + 128 + h * 16, 192, kmax, 16);
        acc = WMMA(a, bb, acc);
      }
      int c = lane & 15, r0 = (lane >> 4) * 8;
#pragma unroll
      for (int v = 0; v < 8; ++v)
        obuf[(mt * 16 + r0 + v) * 64 + h * 16 + c] = acc[v];
    }
    __syncthreads();
  }

  // out projection + residual: X += obuf @ Wo + bo
  for (int t = wv; t < 12; t += 8) {
    int mt = t / 4, nt = t % 4;
    v8f acc = {};
#pragma unroll
    for (int ks = 0; ks < 2; ++ks) {
      v16h a  = fragA(obuf + mt * 16 * 64 + ks * 32, 64, 16, 32);
      v16h bb = fragB(Wo + ks * 32 * 64 + nt * 16, 64, 32, 16);
      acc = WMMA(a, bb, acc);
    }
    int c = lane & 15, r0 = (lane >> 4) * 8;
#pragma unroll
    for (int v = 0; v < 8; ++v) {
      int r = mt * 16 + r0 + v, cc = nt * 16 + c;
      if (r < T) X[(row0 + r) * 64 + cc] += acc[v] + bo[cc];
    }
  }
}

// ---------------------------------------------------------------------------
// 4. Fused FFN over 64-row chunks: X += W2( gelu(W1(tbn(X)) + b1) ) + b2
//    xn and h1 kept in fragment-swizzled LDS (A loads = 2x ds_load_b128).
// ---------------------------------------------------------------------------
__global__ __launch_bounds__(256) void k_ffn(
    float* __restrict__ X,
    const float* __restrict__ meanv, const float* __restrict__ rstdv,
    const float* __restrict__ gam, const float* __restrict__ bet,
    const float* __restrict__ W1, const float* __restrict__ b1,
    const float* __restrict__ W2, const float* __restrict__ b2) {
  __shared__ _Float16 xn[8 * 512]  __attribute__((aligned(32)));   // [mt][ks]
  __shared__ _Float16 h1[16 * 512] __attribute__((aligned(32)));   // [mt][ks0..3]
  int row0 = blockIdx.x * 64;
  int tid = threadIdx.x, lane = tid & 31, wv = tid >> 5;

  for (int i = tid; i < 64 * 64; i += 256) {
    int r = i >> 6, e = i & 63;
    float v = (X[(row0 + r) * 64 + e] - meanv[e]) * rstdv[e] * gam[e] + bet[e];
    xn[((r >> 4) * 2 + (e >> 5)) * 512 + swz(r & 15, e & 31)] = (_Float16)v;
  }
  __syncthreads();

  for (int t = wv; t < 32; t += 8) {           // 4x8 tiles, K=64
    int mt = t >> 3, nt = t & 7;
    v8f acc = {};
#pragma unroll
    for (int ks = 0; ks < 2; ++ks) {
      v16h a  = fragA_swz(xn + (mt * 2 + ks) * 512);
      v16h bb = fragB(W1 + ks * 32 * 128 + nt * 16, 128, 32, 16);
      acc = WMMA(a, bb, acc);
    }
    int c = lane & 15, r0 = (lane >> 4) * 8;
#pragma unroll
    for (int v = 0; v < 8; ++v) {
      int r = mt * 16 + r0 + v, cc = nt * 16 + c;
      h1[((r >> 4) * 4 + (cc >> 5)) * 512 + swz(r & 15, cc & 31)] =
          (_Float16)gelu_erf(acc[v] + b1[cc]);
    }
  }
  __syncthreads();

  for (int t = wv; t < 16; t += 8) {           // 4x4 tiles, K=128
    int mt = t >> 2, nt = t & 3;
    v8f acc = {};
#pragma unroll
    for (int ks = 0; ks < 4; ++ks) {
      v16h a  = fragA_swz(h1 + (mt * 4 + ks) * 512);
      v16h bb = fragB(W2 + ks * 32 * 64 + nt * 16, 64, 32, 16);
      acc = WMMA(a, bb, acc);
    }
    int c = lane & 15, r0 = (lane >> 4) * 8;
#pragma unroll
    for (int v = 0; v < 8; ++v) {
      int r = mt * 16 + r0 + v, cc = nt * 16 + c;
      X[(row0 + r) * 64 + cc] += acc[v] + b2[cc];
    }
  }
}

// ---------------------------------------------------------------------------
// 5. Generic K=64 projection GEMM over 64-row chunks.
//    mode: 0 = f32 store, 1 = f16 store, 2 = f32 residual add
// ---------------------------------------------------------------------------
__global__ __launch_bounds__(256) void k_proj64(
    const float* __restrict__ Xsrc, int normalize,
    const float* __restrict__ meanv, const float* __restrict__ rstdv,
    const float* __restrict__ gam, const float* __restrict__ bet,
    const float* __restrict__ W, int Nall, int col0, int N,
    const float* __restrict__ bias, int mode,
    float* __restrict__ outF, _Float16* __restrict__ outH, int ldo) {
  __shared__ _Float16 xn[8 * 512] __attribute__((aligned(32)));
  int row0 = blockIdx.x * 64;
  int tid = threadIdx.x, lane = tid & 31, wv = tid >> 5;

  for (int i = tid; i < 64 * 64; i += 256) {
    int r = i >> 6, e = i & 63;
    float v = Xsrc[(row0 + r) * 64 + e];
    if (normalize) v = (v - meanv[e]) * rstdv[e] * gam[e] + bet[e];
    xn[((r >> 4) * 2 + (e >> 5)) * 512 + swz(r & 15, e & 31)] = (_Float16)v;
  }
  __syncthreads();

  int ntiles = N >> 4, tiles = 4 * ntiles;
  for (int t = wv; t < tiles; t += 8) {
    int mt = t / ntiles, nt = t % ntiles;
    v8f acc = {};
#pragma unroll
    for (int ks = 0; ks < 2; ++ks) {
      v16h a  = fragA_swz(xn + (mt * 2 + ks) * 512);
      v16h bb = fragB(W + ks * 32 * Nall + col0 + nt * 16, Nall, 32, 16);
      acc = WMMA(a, bb, acc);
    }
    int c = lane & 15, r0 = (lane >> 4) * 8;
#pragma unroll
    for (int v = 0; v < 8; ++v) {
      int r = mt * 16 + r0 + v, cc = nt * 16 + c;
      float z = acc[v] + bias[cc];
      if (mode == 0)      outF[(row0 + r) * ldo + cc] = z;
      else if (mode == 1) outH[(row0 + r) * ldo + cc] = (_Float16)z;
      else                outF[(row0 + r) * ldo + cc] += z;
    }
  }
}

// ---------------------------------------------------------------------------
// 6. Flash cross-attention: one WG per (batch, head); Q[24x16] vs S=2560 keys
//    streamed in 64-key chunks with online softmax; WMMA for QK^T and PV.
// ---------------------------------------------------------------------------
__global__ __launch_bounds__(128) void k_flash(
    const float* __restrict__ Qc, const _Float16* __restrict__ KVh,
    float* __restrict__ Oc, int S) {
  __shared__ float scb[32 * 64];
  __shared__ float alpha[32], lrow[32], mrow[32];
  int b = blockIdx.x >> 2, h = blockIdx.x & 3;
  int tid = threadIdx.x, lane = tid & 31, wv = tid >> 5;

  int mt = wv >> 1;                                   // score M-tile of this wave
  v16h aQ = fragA(Qc + (b * 24 + mt * 16) * 64 + h * 16, 64, 24 - mt * 16, 16);

  if (tid < 32) { mrow[tid] = -1e30f; lrow[tid] = 0.f; }
  v8f oacc = {};                                      // O accumulator (waves 0,1)
  __syncthreads();

  for (int s0 = 0; s0 < S; s0 += 64) {
    // scores: 2 Mtiles x 4 keytiles; wave w does (mt=w>>1, nt=(w&1)*2+{0,1})
#pragma unroll
    for (int j = 0; j < 2; ++j) {
      int nt = (wv & 1) * 2 + j;
      v16h bb = fragA(KVh + (size_t)(b * S + s0 + nt * 16) * 128 + h * 16, 128, 16, 16);
      v8f acc = {};
      acc = WMMA(aQ, bb, acc);
      int c = lane & 15, r0 = (lane >> 4) * 8;
#pragma unroll
      for (int v = 0; v < 8; ++v)
        scb[(mt * 16 + r0 + v) * 64 + nt * 16 + c] = acc[v] * 0.25f;
    }
    __syncthreads();

    if (tid < 32) {                                   // online softmax per row
      int r = tid;
      float* srow = scb + r * 64;
      if (r < 24) {
        float mx = mrow[r];
        for (int c = 0; c < 64; ++c) mx = fmaxf(mx, srow[c]);
        float al = __expf(mrow[r] - mx);
        float l = lrow[r] * al;
        for (int c = 0; c < 64; ++c) { float e = __expf(srow[c] - mx); srow[c] = e; l += e; }
        mrow[r] = mx; lrow[r] = l; alpha[r] = al;
      } else {
        alpha[r] = 1.f;
        for (int c = 0; c < 64; ++c) srow[c] = 0.f;
      }
    }
    __syncthreads();

    if (wv < 2) {                                     // PV: waves 0,1 own O tiles
      int c = lane & 15, r0 = (lane >> 4) * 8;
#pragma unroll
      for (int v = 0; v < 8; ++v) oacc[v] *= alpha[wv * 16 + r0 + v];
#pragma unroll
      for (int ks = 0; ks < 2; ++ks) {
        v16h a  = fragA(scb + wv * 16 * 64 + ks * 32, 64, 16, 32);
        v16h bb = fragB(KVh + (size_t)(b * S + s0 + ks * 32) * 128 + 64 + h * 16, 128, 32, 16);
        oacc = WMMA(a, bb, oacc);
      }
    }
    __syncthreads();
  }

  if (wv < 2) {
    int c = lane & 15, r0 = (lane >> 4) * 8;
#pragma unroll
    for (int v = 0; v < 8; ++v) {
      int r = wv * 16 + r0 + v;
      if (r < 24) Oc[(b * 24 + r) * 64 + h * 16 + c] = oacc[v] / lrow[r];
    }
  }
}

// ---------------------------------------------------------------------------
// 7. d = mem @ aux_out_w + aux_out_b   (one dot of 64 per row)
// ---------------------------------------------------------------------------
__global__ void k_auxout(const float* __restrict__ mem, const float* __restrict__ w,
                         const float* __restrict__ b, float* __restrict__ d, int rows) {
  int r = blockIdx.x * blockDim.x + threadIdx.x;
  if (r >= rows) return;
  float s = 0.f;
  for (int e = 0; e < 64; ++e) s += mem[r * 64 + e] * w[e];
  d[r] = s + b[0];
}

// ---------------------------------------------------------------------------
// 8. expand: dd = gelu(d[64x2560] @ Wexp[2560x5120] + b); 40 WGs x 128 cols
// ---------------------------------------------------------------------------
__global__ __launch_bounds__(256) void k_expand(
    const float* __restrict__ d, const float* __restrict__ Wexp,
    const float* __restrict__ bexp, float* __restrict__ dd) {
  __shared__ _Float16 aT[4 * 512] __attribute__((aligned(32)));
  int n0 = blockIdx.x * 128;
  int tid = threadIdx.x, lane = tid & 31, wv = tid >> 5;
  v8f acc[4] = {{}, {}, {}, {}};

  for (int k0 = 0; k0 < 2560; k0 += 32) {
    __syncthreads();
    for (int i = tid; i < 64 * 32; i += 256) {
      int r = i >> 5, k = i & 31;
      aT[(r >> 4) * 512 + swz(r & 15, k)] = (_Float16)d[r * 2560 + k0 + k];
    }
    __syncthreads();
    if (k0 + 32 < 2560)
      __builtin_prefetch(Wexp + (size_t)(k0 + 32) * 5120 + n0 + wv * 16, 0, 1);
    v16h bb = fragB(Wexp + (size_t)k0 * 5120 + n0 + wv * 16, 5120, 32, 16);
#pragma unroll
    for (int mt = 0; mt < 4; ++mt) {
      v16h a = fragA_swz(aT + mt * 512);
      acc[mt] = WMMA(a, bb, acc[mt]);
    }
  }
  int c = lane & 15, r0 = (lane >> 4) * 8;
  for (int mtv = 0; mtv < 4; ++mtv)
#pragma unroll
    for (int v = 0; v < 8; ++v) {
      int r = mtv * 16 + r0 + v, cc = n0 + wv * 16 + c;
      dd[r * 5120 + cc] = gelu_erf(acc[mtv][v] + bexp[cc]);
    }
}

// ---------------------------------------------------------------------------
// 9. compress: d2 = dd[64x5120] @ Wc[5120x24] + bc (single WG, 8 tiles)
// ---------------------------------------------------------------------------
__global__ __launch_bounds__(256) void k_compress(
    const float* __restrict__ dd, const float* __restrict__ Wc,
    const float* __restrict__ bc, float* __restrict__ d2) {
  __shared__ _Float16 aT[4 * 512] __attribute__((aligned(32)));
  int tid = threadIdx.x, lane = tid & 31, wv = tid >> 5;
  int mt = wv >> 1, nt = wv & 1;
  v8f acc = {};
  for (int k0 = 0; k0 < 5120; k0 += 32) {
    __syncthreads();
    for (int i = tid; i < 64 * 32; i += 256) {
      int r = i >> 5, k = i & 31;
      aT[(r >> 4) * 512 + swz(r & 15, k)] = (_Float16)dd[r * 5120 + k0 + k];
    }
    __syncthreads();
    v16h a  = fragA_swz(aT + mt * 512);
    v16h bb = fragB(Wc + k0 * 24 + nt * 16, 24, 32, 24 - nt * 16);
    acc = WMMA(a, bb, acc);
  }
  int c = lane & 15, r0 = (lane >> 4) * 8;
#pragma unroll
  for (int v = 0; v < 8; ++v) {
    int r = mt * 16 + r0 + v, cc = nt * 16 + c;
    if (cc < 24) d2[r * 24 + cc] = acc[v] + bc[cc];
  }
}

// ---------------------------------------------------------------------------
// 10. t[b,o,:] = d2[b,o]*aux_in_w + aux_in_b + pe[o,:]
// ---------------------------------------------------------------------------
__global__ void k_buildt(const float* __restrict__ d2, const float* __restrict__ aw,
                         const float* __restrict__ ab, const float* __restrict__ pe,
                         float* __restrict__ t, int total) {
  int i = blockIdx.x * blockDim.x + threadIdx.x;
  if (i >= total) return;
  int e = i & 63, ro = i >> 6, o = ro % 24;
  t[i] = d2[ro] * aw[e] + ab[e] + pe[o * 64 + e];
}

// ---------------------------------------------------------------------------
// 11. out[b,o] = t[b,o,:] . out_w + out_b
// ---------------------------------------------------------------------------
__global__ void k_final(const float* __restrict__ t, const float* __restrict__ w,
                        const float* __restrict__ b, float* __restrict__ out, int rows) {
  int r = blockIdx.x * blockDim.x + threadIdx.x;
  if (r >= rows) return;
  float s = 0.f;
  for (int e = 0; e < 64; ++e) s += t[r * 64 + e] * w[e];
  out[r] = s + b[0];
}

// ---------------------------------------------------------------------------
extern "C" void kernel_launch(void* const* d_in, const int* in_sizes, int n_in,
                              void* d_out, int out_size, void* d_ws, size_t ws_size,
                              hipStream_t stream) {
  (void)in_sizes; (void)n_in; (void)out_size; (void)ws_size;
  int i = 0;
  const float* x          = (const float*)d_in[i++];
  const float* in_w       = (const float*)d_in[i++];
  const float* in_b       = (const float*)d_in[i++];
  const float* pe         = (const float*)d_in[i++];
  const float* enc_qkv_w  = (const float*)d_in[i++];
  const float* enc_qkv_b  = (const float*)d_in[i++];
  const float* enc_out_w  = (const float*)d_in[i++];
  const float* enc_out_b  = (const float*)d_in[i++];
  const float* enc_n1_g   = (const float*)d_in[i++];
  const float* enc_n1_b   = (const float*)d_in[i++];
  const float* enc_n2_g   = (const float*)d_in[i++];
  const float* enc_n2_b   = (const float*)d_in[i++];
  const float* enc_ff1_w  = (const float*)d_in[i++];
  const float* enc_ff1_b  = (const float*)d_in[i++];
  const float* enc_ff2_w  = (const float*)d_in[i++];
  const float* enc_ff2_b  = (const float*)d_in[i++];
  const float* dec_sa_qkv_w = (const float*)d_in[i++];
  const float* dec_sa_qkv_b = (const float*)d_in[i++];
  const float* dec_sa_out_w = (const float*)d_in[i++];
  const float* dec_sa_out_b = (const float*)d_in[i++];
  const float* dec_ca_qkv_w = (const float*)d_in[i++];
  const float* dec_ca_qkv_b = (const float*)d_in[i++];
  const float* dec_ca_out_w = (const float*)d_in[i++];
  const float* dec_ca_out_b = (const float*)d_in[i++];
  const float* dec_n1_g   = (const float*)d_in[i++];
  const float* dec_n1_b   = (const float*)d_in[i++];
  const float* dec_n2_g   = (const float*)d_in[i++];
  const float* dec_n2_b   = (const float*)d_in[i++];
  const float* dec_n3_g   = (const float*)d_in[i++];
  const float* dec_n3_b   = (const float*)d_in[i++];
  const float* dec_ff1_w  = (const float*)d_in[i++];
  const float* dec_ff1_b  = (const float*)d_in[i++];
  const float* dec_ff2_w  = (const float*)d_in[i++];
  const float* dec_ff2_b  = (const float*)d_in[i++];
  const float* aux_out_w  = (const float*)d_in[i++];
  const float* aux_out_b  = (const float*)d_in[i++];
  const float* expand_w   = (const float*)d_in[i++];
  const float* expand_b   = (const float*)d_in[i++];
  const float* compress_w = (const float*)d_in[i++];
  const float* compress_b = (const float*)d_in[i++];
  const float* aux_in_w   = (const float*)d_in[i++];
  const float* aux_in_b   = (const float*)d_in[i++];
  const float* out_w      = (const float*)d_in[i++];
  const float* out_b      = (const float*)d_in[i++];

  const int RMEM = 64 * 2560;   // encoder rows
  const int RT   = 64 * 24;     // decoder rows

  char* ws = (char*)d_ws;
  size_t off = 0;
  auto carve = [&](size_t bytes) -> char* {
    char* p = ws + off;
    off = (off + bytes + 255) & ~(size_t)255;
    return p;
  };
  float*     mem     = (float*)carve((size_t)RMEM * 64 * 4);
  float*     partial = (float*)carve(256 * 128 * 4);
  float*     meanb   = (float*)carve(256);
  float*     rstdb   = (float*)carve(256);
  float*     tbuf    = (float*)carve((size_t)RT * 64 * 4);
  float*     Qc      = (float*)carve((size_t)RT * 64 * 4);
  _Float16*  KVh     = (_Float16*)carve((size_t)RMEM * 128 * 2);
  float*     Oc      = (float*)carve((size_t)RT * 64 * 4);
  float*     dbuf    = (float*)carve((size_t)RMEM * 4);
  float*     dd      = (float*)carve((size_t)64 * 5120 * 4);
  float*     d2      = (float*)carve((size_t)64 * 24 * 4);

  // input embed + positional encoding
  k_embed<<<(RMEM * 64 + 255) / 256, 256, 0, stream>>>(x, in_w, in_b, pe, mem, RMEM);

  // ---------------- encoder ----------------
  for (int l = 0; l < 2; ++l) {
    k_stats_partial<<<256, 256, 0, stream>>>(mem, RMEM, 256, partial);
    k_stats_final<<<1, 64, 0, stream>>>(partial, 256, 1.f / RMEM, meanb, rstdb);
    k_attn<<<64 * 64, 256, 0, stream>>>(mem, 64, 40, meanb, rstdb,
        enc_n1_g + l * 64, enc_n1_b + l * 64,
        enc_qkv_w + l * 64 * 192, enc_qkv_b + l * 192,
        enc_out_w + l * 64 * 64, enc_out_b + l * 64);
    k_stats_partial<<<256, 256, 0, stream>>>(mem, RMEM, 256, partial);
    k_stats_final<<<1, 64, 0, stream>>>(partial, 256, 1.f / RMEM, meanb, rstdb);
    k_ffn<<<RMEM / 64, 256, 0, stream>>>(mem, meanb, rstdb,
        enc_n2_g + l * 64, enc_n2_b + l * 64,
        enc_ff1_w + l * 64 * 128, enc_ff1_b + l * 128,
        enc_ff2_w + l * 128 * 64, enc_ff2_b + l * 64);
  }

  // ---------------- bottleneck ----------------
  k_auxout<<<(RMEM + 255) / 256, 256, 0, stream>>>(mem, aux_out_w, aux_out_b, dbuf, RMEM);
  k_expand<<<40, 256, 0, stream>>>(dbuf, expand_w, expand_b, dd);
  k_compress<<<1, 256, 0, stream>>>(dd, compress_w, compress_b, d2);
  k_buildt<<<(RT * 64 + 255) / 256, 256, 0, stream>>>(d2, aux_in_w, aux_in_b, pe, tbuf, RT * 64);

  // ---------------- decoder ----------------
  for (int l = 0; l < 2; ++l) {
    // dense self-attention (T=24, single block per batch)
    k_stats_partial<<<64, 256, 0, stream>>>(tbuf, RT, 64, partial);
    k_stats_final<<<1, 64, 0, stream>>>(partial, 64, 1.f / RT, meanb, rstdb);
    k_attn<<<64, 256, 0, stream>>>(tbuf, 1, 24, meanb, rstdb,
        dec_n1_g + l * 64, dec_n1_b + l * 64,
        dec_sa_qkv_w + l * 64 * 192, dec_sa_qkv_b + l * 192,
        dec_sa_out_w + l * 64 * 64, dec_sa_out_b + l * 64);

    // cross-attention to mem
    k_stats_partial<<<64, 256, 0, stream>>>(tbuf, RT, 64, partial);
    k_stats_final<<<1, 64, 0, stream>>>(partial, 64, 1.f / RT, meanb, rstdb);
    k_proj64<<<RT / 64, 256, 0, stream>>>(tbuf, 1, meanb, rstdb,
        dec_n2_g + l * 64, dec_n2_b + l * 64,
        dec_ca_qkv_w + l * 64 * 192, 192, 0, 64,
        dec_ca_qkv_b + l * 192, 0, Qc, (_Float16*)nullptr, 64);
    k_proj64<<<RMEM / 64, 256, 0, stream>>>(mem, 0, meanb, rstdb,
        (const float*)nullptr, (const float*)nullptr,
        dec_ca_qkv_w + l * 64 * 192, 192, 64, 128,
        dec_ca_qkv_b + l * 192 + 64, 1, (float*)nullptr, KVh, 128);
    k_flash<<<64 * 4, 128, 0, stream>>>(Qc, KVh, Oc, 2560);
    k_proj64<<<RT / 64, 256, 0, stream>>>(Oc, 0, meanb, rstdb,
        (const float*)nullptr, (const float*)nullptr,
        dec_ca_out_w + l * 64 * 64, 64, 0, 64,
        dec_ca_out_b + l * 64, 2, tbuf, (_Float16*)nullptr, 64);

    // FFN
    k_stats_partial<<<64, 256, 0, stream>>>(tbuf, RT, 64, partial);
    k_stats_final<<<1, 64, 0, stream>>>(partial, 64, 1.f / RT, meanb, rstdb);
    k_ffn<<<RT / 64, 256, 0, stream>>>(tbuf, meanb, rstdb,
        dec_n3_g + l * 64, dec_n3_b + l * 64,
        dec_ff1_w + l * 64 * 128, dec_ff1_b + l * 128,
        dec_ff2_w + l * 128 * 64, dec_ff2_b + l * 64);
  }

  // final projection to [B, OUT, 1]
  k_final<<<(RT + 255) / 256, 256, 0, stream>>>(tbuf, out_w, out_b, (float*)d_out, RT);
}
